// BloomAttention_63866163692267
// MI455X (gfx1250) — compile-verified, matrix operand order checked by reading, not run on hardware
//
#include <hip/hip_runtime.h>

typedef __bf16 bf16;
typedef __attribute__((ext_vector_type(16))) __bf16 bf16x16;
typedef __attribute__((ext_vector_type(8)))  float   f32x8;

#define Hh   4096
#define Ss   2048
#define NHh  32
#define HDd  128
#define INV_NORM 0.08838834764831845f   // 1/sqrt(128)

__device__ __forceinline__ f32x8 zero8() {
  f32x8 z;
#pragma unroll
  for (int i = 0; i < 8; ++i) z[i] = 0.0f;
  return z;
}

__device__ __forceinline__ f32x8 wmma_bf16(bf16x16 a, bf16x16 b, f32x8 c) {
  return __builtin_amdgcn_wmma_f32_16x16x32_bf16(false, a, false, b, (short)0, c,
                                                 false, false);
}

// Async DMA: 16B global -> LDS, tracked by ASYNCcnt (no VGPR staging).
// LDS operand = low 32 bits of the generic pointer (flat->LDS truncation).
__device__ __forceinline__ void async_cp16(bf16* l, const bf16* g) {
  const unsigned loff = (unsigned)(unsigned long long)l;
  asm volatile("global_load_async_to_lds_b128 %0, %1, off"
               :: "v"(loff), "v"(g) : "memory");
}

__device__ __forceinline__ void wait_async_zero() {
#if __has_builtin(__builtin_amdgcn_s_wait_asynccnt)
  __builtin_amdgcn_s_wait_asynccnt(0);
#else
  asm volatile("s_wait_asynccnt 0x0" ::: "memory");
#endif
}

// Load a 16x32 bf16 A/B fragment from LDS (row-major, `stride` elements/row).
// Wave32 layout (ISA 7.12.2): lane 0-15 -> row, k in {half*8.., 16+half*8..}
__device__ __forceinline__ bf16x16 load_frag(const bf16* p, int stride, int rowbase,
                                             int k0) {
  const int lane = threadIdx.x & 31;
  const int row  = rowbase + (lane & 15);
  const int half = lane >> 4;
  const bf16* q = p + row * stride + k0 + half * 8;
  bf16x16 f;
#pragma unroll
  for (int i = 0; i < 8; ++i) { f[i] = q[i]; f[i + 8] = q[16 + i]; }
  return f;
}

// ---------------------------------------------------------------------------
// Kernel 1: fused = hidden @ qkv_w^T + qkv_b ; split into Q (pre-scaled),
// K [NH][S][HD], V transposed [NH][HD][S], all bf16. Register-pipelined:
// next k-slab's global loads are issued before the WMMA block.
// grid (96, 16); 256 threads.
// ---------------------------------------------------------------------------
#define TSTR 48
__global__ __launch_bounds__(256) void qkv_gemm_kernel(
    const float* __restrict__ hs, const float* __restrict__ qkv_w,
    const float* __restrict__ qkv_b, bf16* __restrict__ Qb,
    bf16* __restrict__ Kb, bf16* __restrict__ Vt) {
  const int bn = blockIdx.x, bm = blockIdx.y;
  const int tid = threadIdx.x, wave = tid >> 5, lane = tid & 31;

  __shared__ __align__(16) bf16 lA[128 * TSTR];
  __shared__ __align__(16) bf16 lB[128 * TSTR];

  const int mt0 = (wave & 1) * 4;
  const int nt0 = (wave >> 1) * 2;

  f32x8 acc[4][2];
#pragma unroll
  for (int i = 0; i < 4; ++i)
#pragma unroll
    for (int j = 0; j < 2; ++j) acc[i][j] = zero8();

  const float* Aptr = hs + (size_t)(bm * 128) * Hh;
  const float* Bptr = qkv_w + (size_t)(bn * 128) * Hh;
  const int r0 = tid >> 3, c0 = (tid & 7) * 4;

  float4 va[4], vb[4];
  auto load_t = [&](int k0) {
#pragma unroll
    for (int p = 0; p < 4; ++p) {
      const int row = r0 + p * 32;
      va[p] = *(const float4*)(Aptr + (size_t)row * Hh + k0 + c0);
      vb[p] = *(const float4*)(Bptr + (size_t)row * Hh + k0 + c0);
    }
  };
  auto store_t = [&]() {
#pragma unroll
    for (int p = 0; p < 4; ++p) {
      const int row = r0 + p * 32;
      bf16* da = &lA[row * TSTR + c0];
      bf16* db = &lB[row * TSTR + c0];
      da[0] = (bf16)va[p].x; da[1] = (bf16)va[p].y;
      da[2] = (bf16)va[p].z; da[3] = (bf16)va[p].w;
      db[0] = (bf16)vb[p].x; db[1] = (bf16)vb[p].y;
      db[2] = (bf16)vb[p].z; db[3] = (bf16)vb[p].w;
    }
  };

  load_t(0);
  store_t();
  for (int k0 = 0; k0 < Hh; k0 += 32) {
    __syncthreads();                       // tile k0 resident in LDS
    const bool more = (k0 + 32 < Hh);
    if (more) load_t(k0 + 32);             // issue next slab; overlaps WMMAs

    bf16x16 af[4], bfr[2];
#pragma unroll
    for (int i = 0; i < 4; ++i) af[i] = load_frag(lA, TSTR, (mt0 + i) * 16, 0);
#pragma unroll
    for (int j = 0; j < 2; ++j) bfr[j] = load_frag(lB, TSTR, (nt0 + j) * 16, 0);
#pragma unroll
    for (int i = 0; i < 4; ++i)
#pragma unroll
      for (int j = 0; j < 2; ++j) acc[i][j] = wmma_bf16(af[i], bfr[j], acc[i][j]);

    __syncthreads();                       // all waves done reading LDS
    if (more) store_t();                   // convert + stage next tile
  }

  const int which = bn % 3, head = bn / 3;
  const int mbase = (lane >> 4) * 8, nl = lane & 15;
#pragma unroll
  for (int i = 0; i < 4; ++i)
#pragma unroll
    for (int j = 0; j < 2; ++j) {
      const int d = (nt0 + j) * 16 + nl;
      const float bias = qkv_b[bn * 128 + d];
#pragma unroll
      for (int r = 0; r < 8; ++r) {
        const int sg = bm * 128 + (mt0 + i) * 16 + mbase + r;
        const float v = acc[i][j][r] + bias;
        if (which == 0)
          Qb[((size_t)head * Ss + sg) * HDd + d] = (bf16)(v * INV_NORM);
        else if (which == 1)
          Kb[((size_t)head * Ss + sg) * HDd + d] = (bf16)v;
        else
          Vt[((size_t)head * HDd + d) * Ss + sg] = (bf16)v;
      }
    }
}

// ---------------------------------------------------------------------------
// Kernel 2: causal flash attention with ALiBi. grid (NH, S/128); 256 threads.
// Double-buffered K/V tiles staged by GLOBAL_LOAD_ASYNC_TO_LDS_B128: block
// j+1 is DMA'd while block j runs both WMMA phases. P reuses the *current*
// K buffer (disjoint from the buffer being filled).
// ---------------------------------------------------------------------------
#define KSTR 136
#define KVTILE (128 * KSTR)
__global__ __launch_bounds__(256) void attn_kernel(
    const bf16* __restrict__ Qb, const bf16* __restrict__ Kb,
    const bf16* __restrict__ Vt, const float* __restrict__ alibi,
    bf16* __restrict__ ctx) {
  const int head = blockIdx.x, qb = blockIdx.y;
  const int tid = threadIdx.x, wave = tid >> 5, lane = tid & 31;
  const int nl = lane & 15, half = lane >> 4, mbase = half * 8;

  __shared__ __align__(16) bf16 lK[2 * KVTILE];  // current buf reused for P
  __shared__ __align__(16) bf16 lV[2 * KVTILE];

  const bf16* Kh = Kb + (size_t)head * Ss * HDd;
  const bf16* Vh = Vt + (size_t)head * HDd * Ss;
  const float* al = alibi + (size_t)head * Ss;

  // Async-stage K tile [128 keys][128 hd] and V^T tile [128 hd][128 keys].
  auto stage = [&](int jj, int nb) {
    bf16* dK = lK + nb * KVTILE;
    bf16* dV = lV + nb * KVTILE;
#pragma unroll
    for (int p = 0; p < 8; ++p) {
      const int idx = tid + p * 256;       // 2048 16B chunks per array
      const int row = idx >> 4, col8 = (idx & 15) * 8;
      async_cp16(&dK[row * KSTR + col8],
                 Kh + (size_t)(jj * 128 + row) * HDd + col8);
      async_cp16(&dV[row * KSTR + col8],
                 Vh + (size_t)row * Ss + jj * 128 + col8);
    }
  };

  // Q fragments for this wave's 16 rows (HD=128 -> 4 k-chunks), from global.
  bf16x16 qf[4];
  {
    const bf16* qp = Qb + ((size_t)head * Ss + qb * 128 + wave * 16 + nl) * HDd;
#pragma unroll
    for (int kc = 0; kc < 4; ++kc) {
      const bf16* qq = qp + kc * 32 + half * 8;
#pragma unroll
      for (int i = 0; i < 8; ++i) { qf[kc][i] = qq[i]; qf[kc][8 + i] = qq[16 + i]; }
    }
  }

  float rmax[8], rsum[8];
  f32x8 o[8];
#pragma unroll
  for (int r = 0; r < 8; ++r) { rmax[r] = -1e30f; rsum[r] = 0.0f; }
#pragma unroll
  for (int t = 0; t < 8; ++t) o[t] = zero8();

  const int qrow_base = qb * 128 + wave * 16 + mbase;

  stage(0, 0);
  wait_async_zero();
  __syncthreads();

  int buf = 0;
  for (int j = 0; j <= qb; ++j) {
    if (j < qb) stage(j + 1, buf ^ 1);     // DMA next block during compute
    bf16* lKb = lK + buf * KVTILE;
    bf16* lVb = lV + buf * KVTILE;

    // Scores: S = Q K^T  (8 key-tiles x 4 k-chunks WMMAs)
    f32x8 sacc[8];
#pragma unroll
    for (int t = 0; t < 8; ++t) {
      sacc[t] = zero8();
#pragma unroll
      for (int kc = 0; kc < 4; ++kc) {
        bf16x16 b = load_frag(lKb, KSTR, t * 16, kc * 32);
        sacc[t] = wmma_bf16(qf[kc], b, sacc[t]);
      }
    }
    __syncthreads();  // all waves done reading lKb -> safe to reuse for P

    // ALiBi + causal mask (Q was pre-scaled by 1/sqrt(HD))
    const bool diag = (j == qb);
#pragma unroll
    for (int t = 0; t < 8; ++t) {
      const int kg = j * 128 + t * 16 + nl;
      const float a = al[kg];
#pragma unroll
      for (int r = 0; r < 8; ++r) {
        float v = sacc[t][r] + a;
        if (diag && kg > qrow_base + r) v = -1e30f;
        sacc[t][r] = v;
      }
    }

    // Online softmax (row stats replicated across 16-lane groups)
#pragma unroll
    for (int r = 0; r < 8; ++r) {
      float m = rmax[r];
#pragma unroll
      for (int t = 0; t < 8; ++t) m = fmaxf(m, sacc[t][r]);
      m = fmaxf(m, __shfl_xor(m, 1));
      m = fmaxf(m, __shfl_xor(m, 2));
      m = fmaxf(m, __shfl_xor(m, 4));
      m = fmaxf(m, __shfl_xor(m, 8));
      const float alpha = __expf(rmax[r] - m);
      float rs = 0.0f;
#pragma unroll
      for (int t = 0; t < 8; ++t) {
        const float pv = __expf(sacc[t][r] - m);
        sacc[t][r] = pv;
        rs += pv;
      }
      rs += __shfl_xor(rs, 1);
      rs += __shfl_xor(rs, 2);
      rs += __shfl_xor(rs, 4);
      rs += __shfl_xor(rs, 8);
      rsum[r] = rsum[r] * alpha + rs;
      rmax[r] = m;
#pragma unroll
      for (int t = 0; t < 8; ++t) o[t][r] *= alpha;
    }

    // Write P (bf16) into this wave's private slice of the current K buffer.
    bf16* lP = lKb + wave * 16 * KSTR;
#pragma unroll
    for (int t = 0; t < 8; ++t)
#pragma unroll
      for (int r = 0; r < 8; ++r)
        lP[(mbase + r) * KSTR + t * 16 + nl] = (bf16)sacc[t][r];

    bf16x16 pf[4];
#pragma unroll
    for (int kc = 0; kc < 4; ++kc) pf[kc] = load_frag(lP, KSTR, 0, kc * 32);

    // O += P @ V  (8 d-tiles x 4 k-chunks WMMAs)
#pragma unroll
    for (int t = 0; t < 8; ++t)
#pragma unroll
      for (int kc = 0; kc < 4; ++kc) {
        bf16x16 b = load_frag(lVb, KSTR, t * 16, kc * 32);
        o[t] = wmma_bf16(pf[kc], b, o[t]);
      }

    wait_async_zero();   // next block's DMA complete (this wave) ...
    __syncthreads();     // ... and all waves' DMA + LDS reads complete
    buf ^= 1;
  }

  // Normalize and emit ctx [S][H] bf16.
#pragma unroll
  for (int t = 0; t < 8; ++t)
#pragma unroll
    for (int r = 0; r < 8; ++r) {
      const int sg = qrow_base + r;
      const int d = t * 16 + nl;
      ctx[(size_t)sg * Hh + head * HDd + d] = (bf16)(o[t][r] / rsum[r]);
    }
}

// ---------------------------------------------------------------------------
// Kernel 3: out = residual + ctx @ dense_w^T + dense_b. Register-pipelined.
// grid (32, 16); 256 threads.
// ---------------------------------------------------------------------------
__global__ __launch_bounds__(256) void dense_gemm_kernel(
    const bf16* __restrict__ ctx, const float* __restrict__ dw,
    const float* __restrict__ db, const float* __restrict__ res,
    float* __restrict__ out) {
  const int bn = blockIdx.x, bm = blockIdx.y;
  const int tid = threadIdx.x, wave = tid >> 5, lane = tid & 31;

  __shared__ __align__(16) bf16 lA[128 * TSTR];
  __shared__ __align__(16) bf16 lB[128 * TSTR];

  const int mt0 = (wave & 1) * 4;
  const int nt0 = (wave >> 1) * 2;

  f32x8 acc[4][2];
#pragma unroll
  for (int i = 0; i < 4; ++i)
#pragma unroll
    for (int j = 0; j < 2; ++j) acc[i][j] = zero8();

  const bf16* Aptr = ctx + (size_t)(bm * 128) * Hh;
  const float* Bptr = dw + (size_t)(bn * 128) * Hh;
  const int r0 = tid >> 3, c0 = (tid & 7) * 4;

  uint4 ua[2];
  float4 vb[4];
  auto load_t = [&](int k0) {
#pragma unroll
    for (int p = 0; p < 2; ++p) {           // A: 512 16B chunks, 2/thread
      const int idx = tid + p * 256;
      const int row = idx >> 2, col8 = (idx & 3) * 8;
      ua[p] = *(const uint4*)(Aptr + (size_t)row * Hh + k0 + col8);
    }
#pragma unroll
    for (int p = 0; p < 4; ++p) {           // B: fp32
      const int row = r0 + p * 32;
      vb[p] = *(const float4*)(Bptr + (size_t)row * Hh + k0 + c0);
    }
  };
  auto store_t = [&]() {
#pragma unroll
    for (int p = 0; p < 2; ++p) {
      const int idx = tid + p * 256;
      const int row = idx >> 2, col8 = (idx & 3) * 8;
      *(uint4*)&lA[row * TSTR + col8] = ua[p];
    }
#pragma unroll
    for (int p = 0; p < 4; ++p) {
      const int row = r0 + p * 32;
      bf16* d = &lB[row * TSTR + c0];
      d[0] = (bf16)vb[p].x; d[1] = (bf16)vb[p].y;
      d[2] = (bf16)vb[p].z; d[3] = (bf16)vb[p].w;
    }
  };

  load_t(0);
  store_t();
  for (int k0 = 0; k0 < Hh; k0 += 32) {
    __syncthreads();
    const bool more = (k0 + 32 < Hh);
    if (more) load_t(k0 + 32);

    bf16x16 af[4], bfv[2];
#pragma unroll
    for (int i = 0; i < 4; ++i) af[i] = load_frag(lA, TSTR, (mt0 + i) * 16, 0);
#pragma unroll
    for (int j = 0; j < 2; ++j) bfv[j] = load_frag(lB, TSTR, (nt0 + j) * 16, 0);
#pragma unroll
    for (int i = 0; i < 4; ++i)
#pragma unroll
      for (int j = 0; j < 2; ++j) acc[i][j] = wmma_bf16(af[i], bfv[j], acc[i][j]);

    __syncthreads();
    if (more) store_t();
  }

  const int mbase = (lane >> 4) * 8, nl = lane & 15;
#pragma unroll
  for (int i = 0; i < 4; ++i)
#pragma unroll
    for (int j = 0; j < 2; ++j) {
      const int cg = bn * 128 + (nt0 + j) * 16 + nl;
      const float bias = db[cg];
#pragma unroll
      for (int r = 0; r < 8; ++r) {
        const int sg = bm * 128 + (mt0 + i) * 16 + mbase + r;
        out[(size_t)sg * Hh + cg] = res[(size_t)sg * Hh + cg] + acc[i][j][r] + bias;
      }
    }
}

// ---------------------------------------------------------------------------
extern "C" void kernel_launch(void* const* d_in, const int* in_sizes, int n_in,
                              void* d_out, int out_size, void* d_ws,
                              size_t ws_size, hipStream_t stream) {
  (void)in_sizes; (void)n_in; (void)out_size; (void)ws_size;
  const float* hs      = (const float*)d_in[0];
  const float* residual= (const float*)d_in[1];
  const float* alibi   = (const float*)d_in[2];
  // d_in[3] = attention_mask (causality computed analytically)
  const float* qkv_w   = (const float*)d_in[4];
  const float* qkv_b   = (const float*)d_in[5];
  const float* dense_w = (const float*)d_in[6];
  const float* dense_b = (const float*)d_in[7];
  float* out = (float*)d_out;

  const size_t per = (size_t)NHh * Ss * HDd;  // 8M elements
  bf16* Qb  = (bf16*)d_ws;
  bf16* Kb  = Qb + per;
  bf16* Vt  = Kb + per;
  bf16* ctx = Vt + per;

  qkv_gemm_kernel<<<dim3(96, 16), 256, 0, stream>>>(hs, qkv_w, qkv_b, Qb, Kb, Vt);
  attn_kernel<<<dim3(NHh, Ss / 128), 256, 0, stream>>>(Qb, Kb, Vt, alibi, ctx);
  dense_gemm_kernel<<<dim3(32, 16), 256, 0, stream>>>(ctx, dense_w, dense_b,
                                                      residual, out);
}
